// Moser_48550310314294
// MI455X (gfx1250) — matrix operand order repaired and checked by reading, not compiled, not run on hardware
//
#include <hip/hip_runtime.h>
#include <hip/hip_bf16.h>

// ---------------------------------------------------------------------------
// Moser-flow forward: fused MLP + exact divergence (3 forward-mode JVPs).
// [H; T0; T1; T2] stacked -> 128x512 combined tile per 32 samples; the two
// 512x512 hidden layers run as bf16 WMMA GEMMs with activations resident in
// LDS (2 x 130KB ping-pong, CDNA5 320KB WGP LDS). 4x4 output register
// blocking -> x4 fragment reuse (fits LDS/L0 operand rooflines); B fragments
// double-buffered one K-chunk ahead to hide global (L2-hot) load latency.
// ---------------------------------------------------------------------------

typedef __bf16 bf16;
typedef __attribute__((ext_vector_type(16))) __bf16 v16bf;
typedef __attribute__((ext_vector_type(8)))  __bf16 v8bf;
typedef __attribute__((ext_vector_type(8)))  float  v8f;

union V16 { v16bf v; v8bf h[2]; };
union P2  { unsigned int u; __bf16 b[2]; };

#define HID   512
#define SAMP  32             // samples per workgroup
#define MROWS 128            // 4 streams * 32 samples
#define LDST  520            // padded LDS row stride (elements): bank rotation
#define EPS   1e-5f
#define HALF3LOG2PI 2.7568156719207764f   // 0.5 * 3 * log(2*pi)

// ---------------------------------------------------------------------------
// Prep: W[512][512] f32 row-major -> WT[n][k] bf16 (transposed) so B-fragments
// (K-contiguous per output column) are two contiguous 16B loads per lane.
// ---------------------------------------------------------------------------
__global__ void moser_prep(const float* __restrict__ W1, const float* __restrict__ W2,
                           bf16* __restrict__ WT1, bf16* __restrict__ WT2) {
    const int k = blockIdx.x;
    const int n = threadIdx.x;
    if (blockIdx.y == 0) WT1[n * HID + k] = (bf16)W1[k * HID + n];
    else                 WT2[n * HID + k] = (bf16)W2[k * HID + n];
}

// ---------------------------------------------------------------------------
// Fragment helpers (lane layouts per CDNA5 ISA §7.12.2)
// ---------------------------------------------------------------------------
__device__ __forceinline__ void loadA(v16bf a[4], const bf16* __restrict__ in,
                                      int mg, int m, int hf, int kk) {
#pragma unroll
    for (int i = 0; i < 4; ++i) {
        // A 16x32: half0 holds K[0..7],K[16..23]; half1 K[8..15],K[24..31]
        const bf16* pa = in + ((mg * 4 + i) * 16 + m) * LDST + kk * 32 + hf * 8;
        V16 u;
        u.h[0] = *(const v8bf*)(pa);
        u.h[1] = *(const v8bf*)(pa + 16);
        a[i] = u.v;
    }
}

__device__ __forceinline__ void loadB(v16bf b[4], const bf16* __restrict__ WT,
                                      int n0, int m, int hf, int kk) {
#pragma unroll
    for (int j = 0; j < 4; ++j) {
        // B 32x16: lane = column; half0 holds K[0..7],K[16..23]; half1 K[8..15],K[24..31]
        const bf16* pb = WT + ((n0 + j) * 16 + m) * HID + kk * 32 + hf * 8;
        V16 u;
        u.h[0] = *(const v8bf*)(pb);
        u.h[1] = *(const v8bf*)(pb + 16);
        b[j] = u.v;
    }
}

__device__ __forceinline__ void wmma16(v8f acc[4][4], const v16bf a[4], const v16bf b[4]) {
#pragma unroll
    for (int i = 0; i < 4; ++i)
#pragma unroll
        for (int j = 0; j < 4; ++j)
            acc[i][j] = __builtin_amdgcn_wmma_f32_16x16x32_bf16(
                false, a[i], false, b[j], (short)0, acc[i][j], false, false);
}

// ---------------------------------------------------------------------------
// One hidden layer: 128x512 @ 512x512 bf16 GEMM (f32 acc) + nonlinearity.
// ---------------------------------------------------------------------------
__device__ __forceinline__ void layer_pass(const bf16* __restrict__ in,
                                           bf16* __restrict__ out,
                                           const bf16* __restrict__ WT,
                                           const float* __restrict__ bias,
                                           int tid) {
    const int lane = tid & 31;
    const int wave = tid >> 5;
    const int mg   = wave >> 2;       // 0..1: m-blocks mg*4 .. mg*4+3
    const int ng   = wave & 3;        // 0..3: n-group within pass
    const int m    = lane & 15;
    const int hf   = lane >> 4;

#pragma unroll 1
    for (int npass = 0; npass < 2; ++npass) {
        const int n0 = npass * 16 + ng * 4;    // first of this wave's 4 n-tiles
        v8f acc[4][4] = {};

        // One-chunk-ahead double buffering of B (global, L2-hot).
        v16bf b0[4], b1[4];
        loadB(b0, WT, n0, m, hf, 0);
#pragma unroll 1
        for (int kk = 0; kk < 16; kk += 2) {
            v16bf a[4];
            loadB(b1, WT, n0, m, hf, kk + 1);
            loadA(a, in, mg, m, hf, kk);
            wmma16(acc, a, b0);
            loadB(b0, WT, n0, m, hf, (kk + 2) & 15);   // wraps harmlessly at end
            loadA(a, in, mg, m, hf, kk + 1);
            wmma16(acc, a, b1);
        }

        // D layout: VGPR v -> row (hf*8 + v), col = m within each 16x16 tile
#pragma unroll
        for (int i = 0; i < 4; ++i)
#pragma unroll
            for (int j = 0; j < 4; ++j) {
                bf16* po = out + ((mg * 4 + i) * 16 + hf * 8) * LDST + (n0 + j) * 16 + m;
#pragma unroll
                for (int v = 0; v < 8; ++v) po[v * LDST] = (bf16)acc[i][j][v];
            }
    }
    __syncthreads();

    // Elementwise (packed column pairs): H rows -> tanh(z+b);
    // tangent rows -> (1 - h^2) * z_t.
    for (int e = tid; e < SAMP * HID / 2; e += 256) {
        const int s = e >> 8;             // 256 pairs per row
        const int p = (e & 255) * 2;      // even column
        P2 hp; hp.u = *(const unsigned int*)(out + s * LDST + p);
        const float h0 = tanhf((float)hp.b[0] + bias[p]);
        const float h1 = tanhf((float)hp.b[1] + bias[p + 1]);
        const float g0 = 1.0f - h0 * h0;
        const float g1 = 1.0f - h1 * h1;
        hp.b[0] = (bf16)h0; hp.b[1] = (bf16)h1;
        *(unsigned int*)(out + s * LDST + p) = hp.u;
#pragma unroll
        for (int i = 1; i < 4; ++i) {
            bf16* pt = out + (i * SAMP + s) * LDST + p;
            P2 tp; tp.u = *(const unsigned int*)pt;
            tp.b[0] = (bf16)(g0 * (float)tp.b[0]);
            tp.b[1] = (bf16)(g1 * (float)tp.b[1]);
            *(unsigned int*)pt = tp.u;
        }
    }
    __syncthreads();
}

// ---------------------------------------------------------------------------
// Main fused kernel: 1 workgroup = 32 samples, 256 threads (8 wave32).
// ---------------------------------------------------------------------------
__global__ __launch_bounds__(256)
void moser_fused(const float* __restrict__ x,
                 const float* __restrict__ W0, const float* __restrict__ b0,
                 const float* __restrict__ b1, const float* __restrict__ b2,
                 const float* __restrict__ W3,
                 const bf16* __restrict__ WT1, const bf16* __restrict__ WT2,
                 float* __restrict__ outp) {
    extern __shared__ __align__(16) char smem[];
    bf16* bufA = (bf16*)smem;                        // 130KB
    bf16* bufB = (bf16*)(smem + MROWS * LDST * 2);   // 130KB

    const int tid   = threadIdx.x;
    const int lane  = tid & 31;
    const int wave  = tid >> 5;
    const int sbase = blockIdx.x * SAMP;

    // Warm L2 with the layer-2 weights while layer-0/1 run (global_prefetch).
    __builtin_prefetch(WT2 + tid * (HID * HID / 256), 0, 1);

    // ---- Layer 0 (3 -> 512) + tangent seed (packed column pairs)
    for (int e = tid; e < SAMP * HID / 2; e += 256) {
        const int s = e >> 8;
        const int p = (e & 255) * 2;
        const int gs = sbase + s;
        const float x0 = x[gs * 3 + 0], x1 = x[gs * 3 + 1], x2 = x[gs * 3 + 2];
        const float w00 = W0[p],           w01 = W0[p + 1];
        const float w10 = W0[HID + p],     w11 = W0[HID + p + 1];
        const float w20 = W0[2 * HID + p], w21 = W0[2 * HID + p + 1];
        const float h0 = tanhf(x0 * w00 + x1 * w10 + x2 * w20 + b0[p]);
        const float h1 = tanhf(x0 * w01 + x1 * w11 + x2 * w21 + b0[p + 1]);
        const float g0 = 1.0f - h0 * h0;
        const float g1 = 1.0f - h1 * h1;
        P2 v;
        v.b[0] = (bf16)h0;         v.b[1] = (bf16)h1;
        *(unsigned int*)(bufA + s * LDST + p) = v.u;
        v.b[0] = (bf16)(g0 * w00); v.b[1] = (bf16)(g1 * w01);
        *(unsigned int*)(bufA + (SAMP + s) * LDST + p) = v.u;
        v.b[0] = (bf16)(g0 * w10); v.b[1] = (bf16)(g1 * w11);
        *(unsigned int*)(bufA + (2 * SAMP + s) * LDST + p) = v.u;
        v.b[0] = (bf16)(g0 * w20); v.b[1] = (bf16)(g1 * w21);
        *(unsigned int*)(bufA + (3 * SAMP + s) * LDST + p) = v.u;
    }
    __syncthreads();

    // ---- Hidden layers (WMMA GEMMs, ping-pong LDS)
    layer_pass(bufA, bufB, WT1, b1, tid);   // layer 1: bufA -> bufB
    layer_pass(bufB, bufA, WT2, b2, tid);   // layer 2: bufB -> bufA

    // ---- Divergence + output. div = sum_i  T_i . W3[:,i]   (u itself unused)
    // Pair-packed K walk: lane handles K pairs {2*lane + 64*it}.
#pragma unroll 1
    for (int s4 = 0; s4 < 4; ++s4) {
        const int s = wave * 4 + s4;
        float d = 0.0f;
#pragma unroll 1
        for (int it = 0; it < 8; ++it) {
            const int k = 2 * lane + 64 * it;
#pragma unroll
            for (int i = 0; i < 3; ++i) {
                P2 tp; tp.u = *(const unsigned int*)(bufA + ((i + 1) * SAMP + s) * LDST + k);
                d += (float)tp.b[0] * W3[k * 3 + i] + (float)tp.b[1] * W3[(k + 1) * 3 + i];
            }
        }
#pragma unroll
        for (int off = 16; off > 0; off >>= 1) d += __shfl_xor(d, off, 32);
        if (lane == 0) {
            const int gs = sbase + s;
            const float x0 = x[gs * 3 + 0], x1 = x[gs * 3 + 1], x2 = x[gs * 3 + 2];
            const float nu = expf(-0.5f * (x0 * x0 + x1 * x1 + x2 * x2) - HALF3LOG2PI);
            const float sm = nu - d;
            const float mp = fmaxf(sm - EPS, 0.0f) + EPS;
            outp[gs] = -logf(mp);
        }
    }
}

// ---------------------------------------------------------------------------
extern "C" void kernel_launch(void* const* d_in, const int* in_sizes, int n_in,
                              void* d_out, int out_size, void* d_ws, size_t ws_size,
                              hipStream_t stream) {
    const float* x  = (const float*)d_in[0];
    const float* W0 = (const float*)d_in[1];
    const float* b0 = (const float*)d_in[2];
    const float* W1 = (const float*)d_in[3];
    const float* b1 = (const float*)d_in[4];
    const float* W2 = (const float*)d_in[5];
    const float* b2 = (const float*)d_in[6];
    const float* W3 = (const float*)d_in[7];
    // b3 (d_in[8]) does not affect the output: only div(u) and nu are used.

    bf16* WT1 = (bf16*)d_ws;               // 512KB
    bf16* WT2 = WT1 + HID * HID;           // 512KB

    moser_prep<<<dim3(HID, 2), HID, 0, stream>>>(W1, W2, WT1, WT2);

    const int N = in_sizes[0] / 3;               // 65536
    const int nblocks = N / SAMP;                // 2048 workgroups
    const size_t lds = 2u * MROWS * LDST * sizeof(bf16);   // 260KB of 320KB WGP LDS
    moser_fused<<<nblocks, 256, lds, stream>>>(x, W0, b0, b1, b2, W3, WT1, WT2,
                                               (float*)d_out);
}
